// Model_39676907883943
// MI455X (gfx1250) — compile-verified, matrix operand order checked by reading, not run on hardware
//
#include <hip/hip_runtime.h>

typedef float  v8f   __attribute__((ext_vector_type(8)));
typedef float  f4    __attribute__((ext_vector_type(4)));
typedef __bf16 v16bf __attribute__((ext_vector_type(16)));
typedef __bf16 v8bf  __attribute__((ext_vector_type(8)));

#define BATCH 2
#define HEADS 16
#define SEQ   2048
#define DIM   128
#define KBLK  32

// sqrt(128) * log2(e): fold the (multiplicative!) score scale into exp2
#define SCL (11.313708498984761f * 1.4426950408889634f)

__device__ __forceinline__ v8f wmma_bf16(v16bf a, v16bf b, v8f c) {
  return __builtin_amdgcn_wmma_f32_16x16x32_bf16(false, a, false, b, (short)0, c,
                                                 false, false);
}

// 8 fp32 at p -> elements 0..7, 8 fp32 at p+16 -> elements 8..15 (A/B operand halves)
__device__ __forceinline__ v16bf cvt_rows(const float* p) {
  f4 a = *(const f4*)(p);
  f4 b = *(const f4*)(p + 4);
  f4 c = *(const f4*)(p + 16);
  f4 d = *(const f4*)(p + 20);
  v16bf r;
#pragma unroll
  for (int i = 0; i < 4; ++i) {
    r[i]      = (__bf16)a[i];
    r[i + 4]  = (__bf16)b[i];
    r[i + 8]  = (__bf16)c[i];
    r[i + 12] = (__bf16)d[i];
  }
  return r;
}

// two contiguous 16B LDS reads -> one 16-element bf16 operand
__device__ __forceinline__ v16bf lds_tile16(const __bf16* p) {
  v8bf lo = *(const v8bf*)(p);
  v8bf hi = *(const v8bf*)(p + 16);
  return __builtin_shufflevector(lo, hi, 0, 1, 2, 3, 4, 5, 6, 7, 8, 9, 10, 11,
                                 12, 13, 14, 15);
}

__global__ __launch_bounds__(256) void fattn_bf16_wmma(
    const float* __restrict__ Q, const float* __restrict__ K,
    const float* __restrict__ V, const float* __restrict__ Qry,
    const unsigned char* __restrict__ pmask, long long* __restrict__ out) {
  __shared__ __bf16 Klds[KBLK * DIM];   // [k][d], row-major
  __shared__ __bf16 Vtlds[DIM * KBLK];  // [d][k], transposed

  const int lane = threadIdx.x & 31;
  const int wave = threadIdx.x >> 5;
  const int row  = lane & 15;  // query col (B-layout) / M-row (A-layout)
  const int half = lane >> 4;  // lane half selects operand element group
  const int bh   = blockIdx.y;
  const int qbase = (blockIdx.x * 8 + wave) * 16;
  const size_t hoff = (size_t)bh * SEQ * DIM;
  const float* Qh = Q + hoff;
  const float* Kh = K + hoff;
  const float* Vh = V + hoff;

  // Q tile once, in B-operand layout: lane holds its query row, d packed per spec
  v16bf qb[4];
#pragma unroll
  for (int s = 0; s < 4; ++s)
    qb[s] = cvt_rows(Qh + (size_t)(qbase + row) * DIM + 32 * s + half * 8);

  v8f acc[8];
#pragma unroll
  for (int t = 0; t < 8; ++t)
#pragma unroll
    for (int j = 0; j < 8; ++j) acc[t][j] = 0.0f;

  float m_run = -INFINITY;
  float l_run = 0.0f;

  for (int kb = 0; kb < SEQ / KBLK; ++kb) {
    __syncthreads();
    // cooperative stage: K block as bf16 row-major, V block transposed (d-major)
    {
      const int kk = threadIdx.x >> 3;        // 0..31 local key
      const int d0 = (threadIdx.x & 7) * 16;  // 16 d's per thread
      const float* kp = Kh + (size_t)(kb * KBLK + kk) * DIM + d0;
      const float* vp = Vh + (size_t)(kb * KBLK + kk) * DIM + d0;
      f4 k0 = *(const f4*)(kp);     f4 k1 = *(const f4*)(kp + 4);
      f4 k2 = *(const f4*)(kp + 8); f4 k3 = *(const f4*)(kp + 12);
      v8bf h0, h1;
#pragma unroll
      for (int i = 0; i < 4; ++i) {
        h0[i] = (__bf16)k0[i]; h0[4 + i] = (__bf16)k1[i];
        h1[i] = (__bf16)k2[i]; h1[4 + i] = (__bf16)k3[i];
      }
      *(v8bf*)&Klds[kk * DIM + d0]     = h0;
      *(v8bf*)&Klds[kk * DIM + d0 + 8] = h1;
      f4 v0 = *(const f4*)(vp);     f4 v1 = *(const f4*)(vp + 4);
      f4 v2 = *(const f4*)(vp + 8); f4 v3 = *(const f4*)(vp + 12);
#pragma unroll
      for (int i = 0; i < 4; ++i) {
        Vtlds[(d0 + i) * KBLK + kk]      = (__bf16)v0[i];
        Vtlds[(d0 + i + 4) * KBLK + kk]  = (__bf16)v1[i];
        Vtlds[(d0 + i + 8) * KBLK + kk]  = (__bf16)v2[i];
        Vtlds[(d0 + i + 12) * KBLK + kk] = (__bf16)v3[i];
      }
    }
    __syncthreads();

    // S^T = K_blk x Q^T : two 16-key C tiles, contraction over D in 4 slices
    v8f s0, s1;
#pragma unroll
    for (int j = 0; j < 8; ++j) { s0[j] = 0.0f; s1[j] = 0.0f; }
#pragma unroll
    for (int s = 0; s < 4; ++s) {
      v16bf a0 = lds_tile16(&Klds[row * DIM + 32 * s + half * 8]);
      v16bf a1 = lds_tile16(&Klds[(16 + row) * DIM + 32 * s + half * 8]);
      s0 = wmma_bf16(a0, qb[s], s0);
      s1 = wmma_bf16(a1, qb[s], s1);
    }

    // online softmax over the 32 keys of this block (per query column)
    float mx = -INFINITY;
#pragma unroll
    for (int j = 0; j < 8; ++j) mx = fmaxf(mx, fmaxf(s0[j], s1[j]));
    mx = fmaxf(mx, __shfl_xor(mx, 16, 32));
    const float m_new = fmaxf(m_run, mx);
    const float corr = exp2f((m_run - m_new) * SCL);
    float rs = 0.0f;
    v16bf pb;  // P^T already sits in the B-operand layout: just pack to bf16
#pragma unroll
    for (int j = 0; j < 8; ++j) {
      float p0 = exp2f((s0[j] - m_new) * SCL);
      float p1 = exp2f((s1[j] - m_new) * SCL);
      rs += p0 + p1;
      pb[j]     = (__bf16)p0;
      pb[j + 8] = (__bf16)p1;
    }
    rs += __shfl_xor(rs, 16, 32);
    l_run = l_run * corr + rs;
    m_run = m_new;

#pragma unroll
    for (int t = 0; t < 8; ++t)
#pragma unroll
      for (int j = 0; j < 8; ++j) acc[t][j] *= corr;

    // O^T += V^T x P^T  (8 d-slices of 16)
#pragma unroll
    for (int t = 0; t < 8; ++t) {
      v16bf a = lds_tile16(&Vtlds[(16 * t + row) * KBLK + half * 8]);
      acc[t] = wmma_bf16(a, pb, acc[t]);
    }
  }

  // epilogue: normalize, padding-mask override with `query`, truncate to int64
  const float inv_l = 1.0f / l_run;
  const int qidx = qbase + row;
  const bool pm = pmask[(size_t)bh * SEQ + qidx] != 0;
#pragma unroll
  for (int t = 0; t < 8; ++t) {
#pragma unroll
    for (int j = 0; j < 8; ++j) {
      const int d = 16 * t + j + 8 * half;
      float val = acc[t][j] * inv_l;
      if (pm) val = Qry[hoff + (size_t)qidx * DIM + d];
      out[hoff + (size_t)qidx * DIM + d] = (long long)val;
    }
  }
}

extern "C" void kernel_launch(void* const* d_in, const int* in_sizes, int n_in,
                              void* d_out, int out_size, void* d_ws,
                              size_t ws_size, hipStream_t stream) {
  (void)in_sizes; (void)n_in; (void)out_size; (void)d_ws; (void)ws_size;
  const float* q   = (const float*)d_in[0];
  const float* k   = (const float*)d_in[1];
  const float* v   = (const float*)d_in[2];
  const float* qry = (const float*)d_in[3];
  const unsigned char* pm = (const unsigned char*)d_in[4];
  long long* out = (long long*)d_out;

  dim3 grid(SEQ / (16 * 8), BATCH * HEADS);  // 16 x 32 blocks
  dim3 block(256);                           // 8 wave32 waves
  fattn_bf16_wmma<<<grid, block, 0, stream>>>(q, k, v, qry, pm, out);
}
